// BertSelfAttention_2010044694707
// MI455X (gfx1250) — matrix operand order repaired and checked by reading, not compiled, not run on hardware
//
#include <hip/hip_runtime.h>
#include <hip/hip_bf16.h>

// ---------------------------------------------------------------------------
// BERT self-attention forward for MI455X (gfx1250, wave32, WMMA).
//   1) cvt_seq_f16:   seq f32 -> f16
//   2) transpose_w:   W[out,in] f32 -> WT[in,out] f16   (x3)
//   3) qkv_gemm:      LDS-tiled 64x64 workgroup tile, async DMA double-buffer
//                     (global_load_async_to_lds + s_wait_asynccnt), f16 WMMA,
//                     bias fused, Q scaled by 1/sqrt(dv);
//                     Q,V -> [B,H,S,dv], K -> [B,H,dv,S]
//   4) attn:          flash-style online-softmax attention, WMMA f16/f32 acc,
//                     P transposed C-layout -> A-layout through LDS.
// ---------------------------------------------------------------------------

typedef _Float16 v16h __attribute__((ext_vector_type(16)));
typedef _Float16 v8h  __attribute__((ext_vector_type(8)));
typedef float    v8f  __attribute__((ext_vector_type(8)));
typedef int      v4i  __attribute__((ext_vector_type(4)));

#define BATCH  4
#define SEQ    1024
#define DMODEL 1024
#define NHEAD  16
#define DVHEAD 64
#define MROWS  (BATCH * SEQ)   // 4096

// workspace layout (bytes)
#define SEQH_OFF  (0)
#define SEQH_SZ   ((size_t)MROWS * DMODEL * 2)               // 8 MB
#define WT_OFF    (SEQH_OFF + SEQH_SZ)
#define WT_SZ     ((size_t)DMODEL * DMODEL * 2)              // 2 MB each
#define Q_OFF     (WT_OFF + 3 * WT_SZ)
#define QKV_SZ    ((size_t)BATCH * NHEAD * SEQ * DVHEAD * 2) // 8 MB each
#define KT_OFF    (Q_OFF + QKV_SZ)
#define V_OFF     (KT_OFF + QKV_SZ)

static __device__ inline v8f wmma16(v16h a, v16h b, v8f c) {
  // D = A(16x32 f16) * B(32x16 f16) + C(16x16 f32)
  return __builtin_amdgcn_wmma_f32_16x16x32_f16(
      /*neg_a=*/false, a, /*neg_b=*/false, b,
      /*c_mod=*/(short)0, c, /*reuse_a=*/false, /*reuse_b=*/false);
}

static __device__ inline v16h cat8(v8h lo, v8h hi) {
  return __builtin_shufflevector(lo, hi, 0,1,2,3,4,5,6,7,8,9,10,11,12,13,14,15);
}

// 16-byte async DMA global -> LDS (ASYNCcnt path). Parameter types per the
// clang declaration: (int4 AS1*, int4 AS3*, imm offset, imm cpol).
static __device__ inline void async_copy16(const _Float16* g, _Float16* l) {
#if __has_builtin(__builtin_amdgcn_global_load_async_to_lds_b128)
  __builtin_amdgcn_global_load_async_to_lds_b128(
      (__attribute__((address_space(1))) v4i*)(g),
      (__attribute__((address_space(3))) v4i*)(l),
      /*offset=*/0, /*cpol=*/0);
#else
  *(v8h*)l = *(const v8h*)g;
#endif
}

#define WAIT_ASYNC_PREV() asm volatile("s_wait_asynccnt 0x4" ::: "memory")
#define WAIT_ASYNC_ALL()  asm volatile("s_wait_asynccnt 0x0" ::: "memory")

// ---------------------------------------------------------------------------
// 1) seq f32 -> f16
__global__ __launch_bounds__(256)
void cvt_seq_f16(const float* __restrict__ in, _Float16* __restrict__ out, int n) {
  int i = blockIdx.x * 256 + threadIdx.x;
  if (i < n) out[i] = (_Float16)in[i];
}

// 2) WT[k][n] = (f16) W[n][k]   (blockIdx.z selects Wq/Wk/Wv)
__global__ __launch_bounds__(256)
void transpose_w(const float* __restrict__ Wq, const float* __restrict__ Wk,
                 const float* __restrict__ Wv, _Float16* __restrict__ WTq,
                 _Float16* __restrict__ WTk, _Float16* __restrict__ WTv) {
  const float* W = (blockIdx.z == 0) ? Wq : (blockIdx.z == 1) ? Wk : Wv;
  _Float16*  WT = (blockIdx.z == 0) ? WTq : (blockIdx.z == 1) ? WTk : WTv;
  int idx = blockIdx.x * 256 + threadIdx.x;           // idx = k*1024 + n
  int k = idx >> 10, n = idx & 1023;
  WT[idx] = (_Float16)W[n * DMODEL + k];
}

// ---------------------------------------------------------------------------
// 3) QKV projection GEMM, LDS-tiled + async-DMA double-buffered.
//    Workgroup = 128 threads (4 waves) -> 64(M) x 64(N) tile, BK = 32.
//    Per K-step each thread issues 4 async b128 DMAs for the NEXT buffers,
//    waits asynccnt<=4 (previous stage landed), barriers, then feeds WMMA
//    from LDS (ds_load_b128; B fragments broadcast across the 4 waves).
__global__ __launch_bounds__(128)
void qkv_gemm(const _Float16* __restrict__ A,
              const _Float16* __restrict__ WTq, const _Float16* __restrict__ WTk,
              const _Float16* __restrict__ WTv,
              const float* __restrict__ bq, const float* __restrict__ bk,
              const float* __restrict__ bv,
              _Float16* __restrict__ Qh, _Float16* __restrict__ Kt,
              _Float16* __restrict__ Vh) {
  __shared__ _Float16 As[2][64 * 32];   // 4 KB x2
  __shared__ _Float16 Bs[2][32 * 64];   // 4 KB x2

  const int sel = blockIdx.z;
  const _Float16* WT  = (sel == 0) ? WTq : (sel == 1) ? WTk : WTv;
  const float*   bias = (sel == 0) ? bq  : (sel == 1) ? bk  : bv;

  const int tid  = threadIdx.x;         // 0..127
  const int wave = tid >> 5;            // 0..3 -> M sub-tile
  const int lane = tid & 31;
  const int m    = lane & 15;
  const bool hi  = lane >= 16;
  const int mt   = blockIdx.x;          // 0..63  (M tiles of 64)
  const int nt   = blockIdx.y;          // 0..15  (N tiles of 64 == head)

  // DMA one K-stage (64x32 A-tile + 32x64 B-tile) into buffer `buf`.
  auto stage = [&](int kc, int buf) {
#pragma unroll
    for (int p = 0; p < 2; ++p) {       // A tile: 2048 halfs, 8 halfs/thread
      int off = (p * 128 + tid) * 8;
      int row = off >> 5;               // 32 halfs per row
      int col = off & 31;
      async_copy16(A + ((size_t)(mt * 64 + row)) * DMODEL + kc * 32 + col,
                   &As[buf][row * 32 + col]);
    }
#pragma unroll
    for (int p = 0; p < 2; ++p) {       // B tile: 2048 halfs
      int off = (p * 128 + tid) * 8;
      int row = off >> 6;               // 64 halfs per row
      int col = off & 63;
      async_copy16(WT + ((size_t)(kc * 32 + row)) * DMODEL + nt * 64 + col,
                   &Bs[buf][row * 64 + col]);
    }
  };

  v8f acc[4] = {};
  const int c0 = hi ? 8 : 0;
  const int c1 = hi ? 24 : 16;

  stage(0, 0);
  for (int kc = 0; kc < DMODEL / 32; ++kc) {
    const int cur = kc & 1;
    if (kc + 1 < DMODEL / 32) {
      stage(kc + 1, cur ^ 1);
      WAIT_ASYNC_PREV();                // previous stage's 4 DMAs retired
    } else {
      WAIT_ASYNC_ALL();
    }
    __syncthreads();                    // all waves' DMAs visible

    const _Float16* as = &As[cur][(wave * 16 + m) * 32];
    v16h af = cat8(*(const v8h*)(as + c0), *(const v8h*)(as + c1));
    const _Float16* bs = &Bs[cur][m];
#pragma unroll
    for (int f = 0; f < 4; ++f) {
      v16h bf;
#pragma unroll
      for (int s = 0; s < 16; ++s) bf[s] = bs[((s < 8) ? c0 + s : c1 + s - 8) * 64 + f * 16];
      acc[f] = wmma16(af, bf, acc[f]);
    }
    __syncthreads();                    // done reading before buffer reuse
  }

  // epilogue: bias, (1/sqrt(dv)) fold for Q, f16 store into attention layouts
  const float scale = (sel == 0) ? 0.125f : 1.0f;   // 1/sqrt(64)
  const int sg_base = mt * 64 + wave * 16 + (hi ? 8 : 0);
#pragma unroll
  for (int f = 0; f < 4; ++f) {
    const int d  = f * 16 + (lane & 15);
    const float bn = bias[nt * 64 + d];
#pragma unroll
    for (int j = 0; j < 8; ++j) {
      const int sg = sg_base + j;
      const int bb = sg >> 10;          // batch
      const int s  = sg & 1023;         // position
      _Float16 val = (_Float16)((acc[f][j] + bn) * scale);
      if (sel == 1) {                   // K transposed: [B,H,dv,S]
        Kt[(((size_t)bb * NHEAD + nt) * DVHEAD + d) * SEQ + s] = val;
      } else {                          // Q/V: [B,H,S,dv]
        _Float16* dst = (sel == 0) ? Qh : Vh;
        dst[(((size_t)bb * NHEAD + nt) * SEQ + s) * DVHEAD + d] = val;
      }
    }
  }
}

// ---------------------------------------------------------------------------
// 4) Attention. One wave handles (b, h, 16 query rows); streams 32 keys/iter
//    with online softmax. C-fragment row mapping: lanes 0-15 <-> rows 0..7
//    (per VGPR j), lanes 16-31 <-> rows 8..15. Row reductions via shfl_xor
//    masks 1/2/4/8 (stay inside 16-lane halves).
__global__ __launch_bounds__(32)
void attn(const _Float16* __restrict__ Qh, const _Float16* __restrict__ Kt,
          const _Float16* __restrict__ Vh, float* __restrict__ out) {
  __shared__ _Float16 Pl[16 * 32];     // P tile: 16 rows x 32 keys

  const int lane  = threadIdx.x;
  const int m     = lane & 15;
  const bool hi   = lane >= 16;
  const int qt    = blockIdx.x;        // 0..63 query tile
  const int h     = blockIdx.y;
  const int b     = blockIdx.z;
  const int bh    = b * NHEAD + h;
  const int qbase = qt * 16;

  // Q A-fragments (K-dim = dv = 64 -> two 16x32 fragments), scale pre-folded
  v16h aq[2];
  const _Float16* qrow = Qh + ((size_t)bh * SEQ + qbase + m) * DVHEAD;
#pragma unroll
  for (int kk = 0; kk < 2; ++kk) {
    const int q0 = kk * 32 + (hi ? 8 : 0);
    const int q1 = kk * 32 + (hi ? 24 : 16);
    aq[kk] = cat8(*(const v8h*)(qrow + q0), *(const v8h*)(qrow + q1));
  }

  v8f acc[4] = {};                     // out accumulator, 16 x 64 f32
  float mrow[8], lrow[8];
#pragma unroll
  for (int j = 0; j < 8; ++j) { mrow[j] = -1e30f; lrow[j] = 0.0f; }

  const _Float16* ktbase = Kt + (size_t)bh * DVHEAD * SEQ;
  const _Float16* vbase  = Vh + (size_t)bh * SEQ * DVHEAD;

  for (int kc = 0; kc < SEQ / 32; ++kc) {
    const int kb = kc * 32;

    // ---- scores: two 16x16 tiles covering keys kb..kb+31
    v8f s0 = {}, s1 = {};
#pragma unroll
    for (int kk = 0; kk < 2; ++kk) {
      const _Float16* kr = ktbase + kb + m;
      v16h b0, b1;
#pragma unroll
      for (int s = 0; s < 16; ++s) {
        const int d = kk * 32 + ((s < 8) ? (hi ? 8 : 0) + s : (hi ? 24 : 16) + s - 8);
        b0[s] = kr[(size_t)d * SEQ];
        b1[s] = kr[(size_t)d * SEQ + 16];
      }
      s0 = wmma16(aq[kk], b0, s0);
      s1 = wmma16(aq[kk], b1, s1);
    }

    // ---- online softmax over these 32 columns
#pragma unroll
    for (int j = 0; j < 8; ++j) {
      float t = fmaxf(s0[j], s1[j]);
      t = fmaxf(t, __shfl_xor(t, 1, 32));
      t = fmaxf(t, __shfl_xor(t, 2, 32));
      t = fmaxf(t, __shfl_xor(t, 4, 32));
      t = fmaxf(t, __shfl_xor(t, 8, 32));
      const float mnew = fmaxf(mrow[j], t);
      const float corr = __expf(mrow[j] - mnew);
      const float p0 = __expf(s0[j] - mnew);
      const float p1 = __expf(s1[j] - mnew);

      const int row = j + (hi ? 8 : 0);
      Pl[row * 32 + m]      = (_Float16)p0;
      Pl[row * 32 + 16 + m] = (_Float16)p1;

      float ps = p0 + p1;
      ps += __shfl_xor(ps, 1, 32);
      ps += __shfl_xor(ps, 2, 32);
      ps += __shfl_xor(ps, 4, 32);
      ps += __shfl_xor(ps, 8, 32);
      lrow[j] = lrow[j] * corr + ps;
      mrow[j] = mnew;
#pragma unroll
      for (int f = 0; f < 4; ++f) acc[f][j] *= corr;
    }
    __syncthreads();   // P writes visible (DScnt flush)

    // ---- re-read P in A-fragment layout (C->A transpose via LDS)
    const int c0 = hi ? 8 : 0;
    const int c1 = hi ? 24 : 16;
    v16h ap = cat8(*(const v8h*)(&Pl[m * 32 + c0]),
                   *(const v8h*)(&Pl[m * 32 + c1]));
    __syncthreads();   // done reading before next iteration overwrites

    // ---- out += P(16x32) x V(32x64)
    const _Float16* vr = vbase + (size_t)kb * DVHEAD + m;
    __builtin_prefetch(vr + 32 * DVHEAD, 0, 3);   // next 32-key V chunk
#pragma unroll
    for (int f = 0; f < 4; ++f) {
      v16h bv;
#pragma unroll
      for (int s = 0; s < 16; ++s) bv[s] = vr[(size_t)((s < 8) ? c0 + s : c1 + s - 8) * DVHEAD + f * 16];
      acc[f] = wmma16(ap, bv, acc[f]);
    }
  }

  // ---- normalize and store (output layout [B, S, H*dv] fp32)
  const int rbase = qbase + (hi ? 8 : 0);
#pragma unroll
  for (int j = 0; j < 8; ++j) {
    const float inv = 1.0f / lrow[j];
    float* orow = out + ((size_t)b * SEQ + rbase + j) * DMODEL + h * DVHEAD;
#pragma unroll
    for (int f = 0; f < 4; ++f)
      orow[f * 16 + (lane & 15)] = acc[f][j] * inv;
  }
}

// ---------------------------------------------------------------------------
extern "C" void kernel_launch(void* const* d_in, const int* in_sizes, int n_in,
                              void* d_out, int out_size, void* d_ws, size_t ws_size,
                              hipStream_t stream) {
  const float* seq = (const float*)d_in[0];
  const float* Wq  = (const float*)d_in[1];
  const float* bq  = (const float*)d_in[2];
  const float* Wk  = (const float*)d_in[3];
  const float* bk  = (const float*)d_in[4];
  const float* Wv  = (const float*)d_in[5];
  const float* bv  = (const float*)d_in[6];
  float* outp = (float*)d_out;

  char* ws = (char*)d_ws;
  _Float16* seqh = (_Float16*)(ws + SEQH_OFF);
  _Float16* wtq  = (_Float16*)(ws + WT_OFF);
  _Float16* wtk  = (_Float16*)(ws + WT_OFF + WT_SZ);
  _Float16* wtv  = (_Float16*)(ws + WT_OFF + 2 * WT_SZ);
  _Float16* Qhp  = (_Float16*)(ws + Q_OFF);
  _Float16* Ktp  = (_Float16*)(ws + KT_OFF);
  _Float16* Vhp  = (_Float16*)(ws + V_OFF);

  cvt_seq_f16<<<dim3((MROWS * DMODEL) / 256), 256, 0, stream>>>(
      seq, seqh, MROWS * DMODEL);
  transpose_w<<<dim3((DMODEL * DMODEL) / 256, 1, 3), 256, 0, stream>>>(
      Wq, Wk, Wv, wtq, wtk, wtv);
  qkv_gemm<<<dim3(MROWS / 64, NHEAD, 3), 128, 0, stream>>>(
      seqh, wtq, wtk, wtv, bq, bk, bv, Qhp, Ktp, Vhp);
  attn<<<dim3(SEQ / 16, NHEAD, BATCH), 32, 0, stream>>>(
      Qhp, Ktp, Vhp, outp);
}
